// Encoder_overall_ced_3M_68066641707481
// MI455X (gfx1250) — matrix operand order repaired and checked by reading, not compiled, not run on hardware
//
#include <hip/hip_runtime.h>
#include <hip/hip_bf16.h>

// ---------------------------------------------------------------------------
// Types for CDNA5 WMMA (wave32): bf16 A/B fragments (16 x bf16 = 8 VGPRs),
// f32 accumulator (8 x f32 = 8 VGPRs).
// ---------------------------------------------------------------------------
typedef __attribute__((ext_vector_type(16))) __bf16          v16bf;
typedef __attribute__((ext_vector_type(8)))  float           v8f;
typedef __attribute__((ext_vector_type(8)))  unsigned short  ushort8v;  // 16 B
typedef __attribute__((ext_vector_type(4)))  unsigned short  ushort4v;  //  8 B

union Frag16 { ushort8v h[2]; v16bf v; };

__device__ __forceinline__ unsigned short f2bf(float f) {
  // round-to-nearest-even fp32 -> bf16
  unsigned u = __builtin_bit_cast(unsigned, f);
  u += 0x7FFFu + ((u >> 16) & 1u);
  return (unsigned short)(u >> 16);
}

#define BM 128
#define BN 64
#define BK 32
#define LDP 40   // LDS row stride in shorts: 80 B = 16B-aligned, 20-bank step

// C[M,N] partial (f32) = A[M, kStart:kStop] @ B[kStart:kStop, N]
// A is f32 (optionally fused w0*A + w1*A2 + b), B is f32; bf16 conversion on
// the fly in LDS; compute via v_wmma_f32_16x16x32_bf16.
// Block: 256 threads = 8 waves, each wave owns a 16-row stripe of a 128x64 tile.
// blockIdx.z selects a K split; partial written to C + z*M*N.
template<bool FUSE_A>
__global__ __launch_bounds__(256)
void wmma_gemm_kernel(const float* __restrict__ A, const float* __restrict__ A2,
                      const float* __restrict__ wconv, const float* __restrict__ bconv,
                      const float* __restrict__ B, float* __restrict__ C,
                      int M, int N, int K, int kLen)
{
  __shared__ __align__(16) unsigned short As [BM][LDP];  // row-major  (m, k)
  __shared__ __align__(16) unsigned short BsT[BN][LDP];  // transposed (n, k)

  const int tid   = threadIdx.x;
  const int lane  = tid & 31;
  const int wave  = tid >> 5;                 // 0..7 -> 16 M-rows each
  const int mBase = blockIdx.y * BM;
  const int nBase = blockIdx.x * BN;

  int kStart = blockIdx.z * kLen;
  int kStop  = kStart + kLen; if (kStop > K) kStop = K;

  float w0 = 0.f, w1 = 0.f, bc = 0.f;
  if (FUSE_A) { w0 = wconv[0]; w1 = wconv[1]; bc = bconv[0]; }

  v8f acc[4] = {};

  for (int k0 = kStart; k0 < kStop; k0 += BK) {
    // ---- stage A tile (BM x BK): float4 global loads -> packed bf16x4 ------
    for (int i = tid; i < BM * (BK / 4); i += 256) {
      int r  = i >> 3;
      int k4 = (i & 7) * 4;                   // K % 4 == 0 for all our shapes
      ushort4v p = {0, 0, 0, 0};
      if (k0 + k4 < K) {
        size_t gi = (size_t)(mBase + r) * (size_t)K + (size_t)(k0 + k4);
        float4 a = *(const float4*)(A + gi);
        if (FUSE_A) {
          float4 a2 = *(const float4*)(A2 + gi);
          a.x = w0 * a.x + w1 * a2.x + bc;  a.y = w0 * a.y + w1 * a2.y + bc;
          a.z = w0 * a.z + w1 * a2.z + bc;  a.w = w0 * a.w + w1 * a2.w + bc;
        }
        p[0] = f2bf(a.x); p[1] = f2bf(a.y); p[2] = f2bf(a.z); p[3] = f2bf(a.w);
      }
      *(ushort4v*)&As[r][k4] = p;
    }
    // ---- stage B tile transposed (BN cols x BK rows) -----------------------
    // thread loads 4 consecutive K-rows of one column (coalesced across tids),
    // packs into one ds_store_b64 at BsT[col][r4]
    for (int i = tid; i < BN * (BK / 4); i += 256) {
      int c   = i & 63;
      int r4  = (i >> 6) * 4;
      int col = nBase + c;
      ushort4v p = {0, 0, 0, 0};
      #pragma unroll
      for (int j = 0; j < 4; ++j) {
        int kk = k0 + r4 + j;
        float v = (kk < K && col < N) ? B[(size_t)kk * (size_t)N + (size_t)col] : 0.f;
        p[j] = f2bf(v);
      }
      *(ushort4v*)&BsT[c][r4] = p;
    }
    __syncthreads();

    // ---- preload ALL fragments first (pipeline the DS unit), then issue ----
    // ---- four back-to-back WMMAs (single dscnt wait, XDL stays busy) -------
    // A: lanes 0-15: M=lane, K runs [0,8)+[16,24); lanes16-31: M=lane-16,
    //    K runs [8,16)+[24,32). B: column = lane&15, same K runs as A.
    const int m     = lane & 15;
    const int kh    = lane >> 4;
    const int row   = wave * 16 + m;
    const int kBase = kh * 16;

    Frag16 fa;
    fa.h[0] = *(const ushort8v*)&As[row][kh * 8];
    fa.h[1] = *(const ushort8v*)&As[row][16 + kh * 8];

    Frag16 fb[4];
    #pragma unroll
    for (int g = 0; g < 4; ++g) {
      fb[g].h[0] = *(const ushort8v*)&BsT[g * 16 + m][kh * 8];
      fb[g].h[1] = *(const ushort8v*)&BsT[g * 16 + m][16 + kh * 8];
    }

    #pragma unroll
    for (int g = 0; g < 4; ++g) {
      acc[g] = __builtin_amdgcn_wmma_f32_16x16x32_bf16(
          /*neg_a=*/false, fa.v, /*neg_b=*/false, fb[g].v,
          /*c_mod=*/(short)0, acc[g], /*reuse_a=*/false, /*reuse_b=*/false);
    }
    __syncthreads();
  }

  // ---- store partial: VGPR r -> M=r (lanes 0-15) / r+8 (lanes 16-31) -------
  float* Cz = C + (size_t)blockIdx.z * (size_t)M * (size_t)N;
  const int n  = lane & 15;
  const int mo = (lane >> 4) * 8;
  #pragma unroll
  for (int g = 0; g < 4; ++g) {
    int col = nBase + g * 16 + n;
    #pragma unroll
    for (int r = 0; r < 8; ++r) {
      if (col < N)
        Cz[(size_t)(mBase + wave * 16 + mo + r) * (size_t)N + (size_t)col] = acc[g][r];
    }
  }
}

// out[i] = sum_z part[z*MN + i]  (deterministic split-K reduction)
__global__ __launch_bounds__(256)
void reduce_split_kernel(const float* __restrict__ part, float* __restrict__ out,
                         int MN, int S)
{
  int i = blockIdx.x * 256 + threadIdx.x;
  if (i < MN) {
    float s = 0.f;
    for (int z = 0; z < S; ++z) s += part[(size_t)z * (size_t)MN + i];
    out[i] = s;
  }
}

// ---------------------------------------------------------------------------
// CED: emb = gc + alpha * ( relu(LN(gc) @ W1 + b1) @ W2 + b2 )
// ---------------------------------------------------------------------------
__global__ __launch_bounds__(64)
void ced_kernel(const float* __restrict__ gc,
                const float* __restrict__ g, const float* __restrict__ be,
                const float* __restrict__ W1, const float* __restrict__ b1,
                const float* __restrict__ W2, const float* __restrict__ b2,
                const float* __restrict__ alphaPtr, float* __restrict__ emb)
{
  const int row = blockIdx.x;
  const int t   = threadIdx.x;          // 0..63
  __shared__ float red[64];
  __shared__ float y[64];
  __shared__ float h[32];

  float x = gc[(size_t)row * 64 + t];

  red[t] = x; __syncthreads();
  for (int s = 32; s > 0; s >>= 1) { if (t < s) red[t] += red[t + s]; __syncthreads(); }
  float mu = red[0] * (1.0f / 64.0f);
  __syncthreads();

  float d = x - mu;
  red[t] = d * d; __syncthreads();
  for (int s = 32; s > 0; s >>= 1) { if (t < s) red[t] += red[t + s]; __syncthreads(); }
  float var = red[0] * (1.0f / 64.0f);
  __syncthreads();

  y[t] = d * __frsqrt_rn(var + 1e-5f) * g[t] + be[t];
  __syncthreads();

  if (t < 32) {
    float s = b1[t];
    #pragma unroll 8
    for (int k = 0; k < 64; ++k) s = fmaf(y[k], W1[k * 32 + t], s);
    h[t] = fmaxf(s, 0.0f);
  }
  __syncthreads();

  float e = b2[t];
  #pragma unroll 8
  for (int j = 0; j < 32; ++j) e = fmaf(h[j], W2[j * 64 + t], e);
  emb[(size_t)row * 64 + t] = x + alphaPtr[0] * e;
}

// ---------------------------------------------------------------------------
// Fusion MLP: comb = ([emb1|emb2|emb3] @ Wf1 + bf1) @ Wf2 + bf2
// ---------------------------------------------------------------------------
__global__ __launch_bounds__(64)
void fusion_kernel(const float* __restrict__ e1, const float* __restrict__ e2,
                   const float* __restrict__ e3,
                   const float* __restrict__ Wf1, const float* __restrict__ bf1,
                   const float* __restrict__ Wf2, const float* __restrict__ bf2,
                   float* __restrict__ comb)
{
  const int row = blockIdx.x;
  const int t   = threadIdx.x;          // 0..63
  __shared__ float c[192];
  __shared__ float h[64];

  c[t]       = e1[(size_t)row * 64 + t];
  c[64 + t]  = e2[(size_t)row * 64 + t];
  c[128 + t] = e3[(size_t)row * 64 + t];
  __syncthreads();

  float s = bf1[t];
  #pragma unroll 8
  for (int k = 0; k < 192; ++k) s = fmaf(c[k], Wf1[k * 64 + t], s);
  h[t] = s;
  __syncthreads();

  float o = bf2[t];
  #pragma unroll 8
  for (int j = 0; j < 64; ++j) o = fmaf(h[j], Wf2[j * 64 + t], o);
  comb[(size_t)row * 64 + t] = o;
}

// ---------------------------------------------------------------------------
// Host side
// ---------------------------------------------------------------------------
extern "C" void kernel_launch(void* const* d_in, const int* in_sizes, int n_in,
                              void* d_out, int out_size, void* d_ws, size_t ws_size,
                              hipStream_t stream)
{
  (void)in_sizes; (void)n_in; (void)out_size; (void)ws_size;

  const float* feat[3]  = {(const float*)d_in[0], (const float*)d_in[1], (const float*)d_in[2]};
  const float* adjsp[3] = {(const float*)d_in[3], (const float*)d_in[5], (const float*)d_in[7]};
  const float* adjft[3] = {(const float*)d_in[4], (const float*)d_in[6], (const float*)d_in[8]};

  const int base[3] = {9, 20, 31};
  const float *wconv[3], *bconv[3], *Wenc[3], *gced[3], *bece[3];
  const float *W1[3], *b1[3], *W2[3], *b2[3], *alpha[3], *Wdec[3];
  for (int i = 0; i < 3; ++i) {
    int b = base[i];
    wconv[i] = (const float*)d_in[b + 0];
    bconv[i] = (const float*)d_in[b + 1];
    Wenc[i]  = (const float*)d_in[b + 2];
    gced[i]  = (const float*)d_in[b + 3];
    bece[i]  = (const float*)d_in[b + 4];
    W1[i]    = (const float*)d_in[b + 5];
    b1[i]    = (const float*)d_in[b + 6];
    W2[i]    = (const float*)d_in[b + 7];
    b2[i]    = (const float*)d_in[b + 8];
    alpha[i] = (const float*)d_in[b + 9];
    Wdec[i]  = (const float*)d_in[b + 10];
  }
  const float* Wf1 = (const float*)d_in[42];
  const float* bf1 = (const float*)d_in[43];
  const float* Wf2 = (const float*)d_in[44];
  const float* bf2 = (const float*)d_in[45];

  const int NN = 4096, DOUT = 64, SPLIT = 8;
  const int Dx[3] = {3000, 1000, 500};
  const int MN64  = NN * DOUT;                  // 262144

  // workspace: fe[3], gc[3] (f32 [N,64]), dec[3] (f32 [N,Di]), splitK partials
  float* ws = (float*)d_ws;
  float* fe[3] = { ws,                          ws + (size_t)MN64,     ws + 2 * (size_t)MN64 };
  float* gc[3] = { ws + 3 * (size_t)MN64, ws + 4 * (size_t)MN64, ws + 5 * (size_t)MN64 };
  float* dec[3];
  dec[0] = ws + 6 * (size_t)MN64;
  dec[1] = dec[0] + (size_t)NN * Dx[0];
  dec[2] = dec[1] + (size_t)NN * Dx[1];
  float* part = dec[2] + (size_t)NN * Dx[2];    // SPLIT * MN64 floats

  // outputs flat in return order: emb1, emb2, emb3, comb, rec1, rec2, rec3
  float* out = (float*)d_out;
  float* emb[3] = { out, out + (size_t)MN64, out + 2 * (size_t)MN64 };
  float* comb   = out + 3 * (size_t)MN64;
  float* rec[3];
  rec[0] = out + 4 * (size_t)MN64;
  rec[1] = rec[0] + (size_t)NN * Dx[0];
  rec[2] = rec[1] + (size_t)NN * Dx[1];

  dim3 blk(256);
  dim3 redGrid((MN64 + 255) / 256);

  for (int i = 0; i < 3; ++i) {
    // fe = feat @ W_enc   [4096 x 64], K = Di  -- split-K x8 for parallelism
    int kLenFe = ((Dx[i] + SPLIT * BK - 1) / (SPLIT * BK)) * BK;
    wmma_gemm_kernel<false><<<dim3(1, NN / BM, SPLIT), blk, 0, stream>>>(
        feat[i], nullptr, nullptr, nullptr, Wenc[i], part, NN, DOUT, Dx[i], kLenFe);
    reduce_split_kernel<<<redGrid, blk, 0, stream>>>(part, fe[i], MN64, SPLIT);

    // gc = (w0*adj_sp + w1*adj_ft + b) @ fe   [4096 x 64], K = 4096, fused conv
    wmma_gemm_kernel<true><<<dim3(1, NN / BM, SPLIT), blk, 0, stream>>>(
        adjsp[i], adjft[i], wconv[i], bconv[i], fe[i], part, NN, DOUT, NN, NN / SPLIT);
    reduce_split_kernel<<<redGrid, blk, 0, stream>>>(part, gc[i], MN64, SPLIT);

    // emb = CED(gc)
    ced_kernel<<<NN, 64, 0, stream>>>(gc[i], gced[i], bece[i],
                                      W1[i], b1[i], W2[i], b2[i], alpha[i], emb[i]);
  }

  fusion_kernel<<<NN, 64, 0, stream>>>(emb[0], emb[1], emb[2], Wf1, bf1, Wf2, bf2, comb);

  for (int i = 0; i < 3; ++i) {
    dim3 g((Dx[i] + BN - 1) / BN, NN / BM, 1);
    // dec = comb @ W_dec   [4096 x Di], K = 64
    wmma_gemm_kernel<false><<<g, blk, 0, stream>>>(
        comb, nullptr, nullptr, nullptr, Wdec[i], dec[i], NN, Dx[i], DOUT, DOUT);
    // rec = adj_sp @ dec   [4096 x Di], K = 4096  (the 151 GF hot loop)
    wmma_gemm_kernel<false><<<g, blk, 0, stream>>>(
        adjsp[i], nullptr, nullptr, nullptr, dec[i], rec[i], NN, Dx[i], NN, NN);
  }
}
